// LanguageModel_63617055588887
// MI455X (gfx1250) — compile-verified, matrix operand order checked by reading, not run on hardware
//
#include <hip/hip_runtime.h>
#include <hip/hip_bf16.h>

// Problem constants (match reference)
#define Bc   4
#define Sc   1024
#define Dc   1024
#define Hc   16
#define Lc   4
#define Vc   32000
#define DHc  64
#define DFFc 4096
#define Mc   (Bc * Sc)   // 4096 token rows

typedef __attribute__((ext_vector_type(16))) __bf16 v16bf;
typedef __attribute__((ext_vector_type(8)))  __bf16 v8bf;
typedef __attribute__((ext_vector_type(8)))  float  v8f;

union Frag16 { v16bf v; v8bf p[2]; };

enum { FLAG_BIAS = 1, FLAG_RES = 2, FLAG_RELU = 4, FLAG_OUTF32 = 8, FLAG_OUTBF16 = 16 };

// ---------------------------------------------------------------------------
// Embedding gather + positional add (f32)
// ---------------------------------------------------------------------------
__global__ __launch_bounds__(256)
void embed_kernel(const int* __restrict__ tok, const float* __restrict__ E,
                  const float* __restrict__ P, float* __restrict__ X)
{
    size_t i = (size_t)blockIdx.x * 256 + threadIdx.x;   // over M*D
    int d  = (int)(i & (Dc - 1));
    size_t bs = i >> 10;                                  // D = 1024
    int s  = (int)(bs & (Sc - 1));
    int tk = tok[bs];
    X[i] = E[(size_t)tk * Dc + d] + P[(size_t)s * Dc + d];
}

// ---------------------------------------------------------------------------
// Row LayerNorm: f32 in -> bf16 out (one block per row, D = 1024)
// ---------------------------------------------------------------------------
__global__ __launch_bounds__(256)
void ln_kernel(const float* __restrict__ X, const float* __restrict__ g,
               const float* __restrict__ b, __bf16* __restrict__ out)
{
    const int row = blockIdx.x, t = threadIdx.x;
    const float* x = X + (size_t)row * Dc;
    float v[4], s = 0.f, ss = 0.f;
#pragma unroll
    for (int j = 0; j < 4; j++) { v[j] = x[t + j * 256]; s += v[j]; ss += v[j] * v[j]; }
    __shared__ float rs[256], rq[256];
    rs[t] = s; rq[t] = ss; __syncthreads();
    for (int off = 128; off > 0; off >>= 1) {
        if (t < off) { rs[t] += rs[t + off]; rq[t] += rq[t + off]; }
        __syncthreads();
    }
    const float mean = rs[0] * (1.0f / Dc);
    const float var  = rq[0] * (1.0f / Dc) - mean * mean;
    const float rstd = rsqrtf(var + 1e-5f);
#pragma unroll
    for (int j = 0; j < 4; j++) {
        int c = t + j * 256;
        out[(size_t)row * Dc + c] = (__bf16)((v[j] - mean) * rstd * g[c] + b[c]);
    }
}

// ---------------------------------------------------------------------------
// f32 -> bf16 straight conversion (contiguous)
// ---------------------------------------------------------------------------
__global__ __launch_bounds__(256)
void cvt_kernel(const float* __restrict__ in, __bf16* __restrict__ out, int n)
{
    int i = blockIdx.x * 256 + threadIdx.x;
    if (i < n) out[i] = (__bf16)in[i];
}

// ---------------------------------------------------------------------------
// Tiled transpose + f32->bf16 convert: out[c*ldo + r] = in[r*ldi + c]
// Batched via blockIdx.z with simple byte strides (elements).
// Used to build all GEMM B-operands pre-transposed into [N][K] layout.
// ---------------------------------------------------------------------------
__global__ __launch_bounds__(256)
void transpose_cvt_kernel(const float* __restrict__ in, __bf16* __restrict__ out,
                          int ldi, int ldo, long inb, long outb)
{
    __shared__ float tile[32][33];
    const float* ip = in  + (size_t)blockIdx.z * inb;
    __bf16*      op = out + (size_t)blockIdx.z * outb;
    const int r0 = blockIdx.y * 32, c0 = blockIdx.x * 32;
    const int tx = threadIdx.x & 31, ty = threadIdx.x >> 5;   // 32x8
#pragma unroll
    for (int j = 0; j < 32; j += 8)
        tile[ty + j][tx] = ip[(size_t)(r0 + ty + j) * ldi + c0 + tx];
    __syncthreads();
#pragma unroll
    for (int j = 0; j < 32; j += 8)
        op[(size_t)(c0 + ty + j) * ldo + r0 + tx] = (__bf16)tile[tx][ty + j];
}

// ---------------------------------------------------------------------------
// V [B*S, D] bf16 -> Vt [B,H,DH,S] bf16 (tiled, coalesced both sides)
// ---------------------------------------------------------------------------
__global__ __launch_bounds__(256)
void transpose_v_kernel(const __bf16* __restrict__ Vb, __bf16* __restrict__ Vt)
{
    __shared__ __bf16 tile[32][34];
    const int bh = blockIdx.z, h = bh & 15, b = bh >> 4;
    const int s0 = blockIdx.y * 32, e0 = blockIdx.x * 32;
    const int tx = threadIdx.x & 31, ty = threadIdx.x >> 5;
    const __bf16* ip = Vb + (size_t)b * Sc * Dc + h * DHc;
    __bf16*       op = Vt + (size_t)bh * DHc * Sc;
#pragma unroll
    for (int j = 0; j < 32; j += 8)
        tile[ty + j][tx] = ip[(size_t)(s0 + ty + j) * Dc + e0 + tx];
    __syncthreads();
#pragma unroll
    for (int j = 0; j < 32; j += 8)
        op[(size_t)(e0 + ty + j) * Sc + s0 + tx] = tile[tx][ty + j];
}

// ---------------------------------------------------------------------------
// bf16 WMMA GEMM with pre-transposed B: C[M,N] = A[M,K] * Bt[N,K]^T
// 128x128 block tile, 8 waves, each wave 64x32 (4x2 wmma tiles), K step 64.
// Both A and Bt tiles stage into LDS with b128 stores; fragments are
// ds_load_b128 pairs matching the 16-bit WMMA A/B VGPR layout.
// ---------------------------------------------------------------------------
__global__ __launch_bounds__(256, 2)
void gemm_kernel(const __bf16* __restrict__ A, const __bf16* __restrict__ Bt,
                 const float* __restrict__ bias, const float* __restrict__ Res,
                 float* __restrict__ Cf, __bf16* __restrict__ Cb,
                 int M, int N, int K, int flags)
{
    __shared__ __bf16 sA[128][72];   // rows = 144B (16B aligned, bank-skewed)
    __shared__ __bf16 sB[128][72];   // sB[n][k]
    const int t    = threadIdx.x;
    const int lane = t & 31, wave = t >> 5;
    const int wm = wave >> 2, wn = wave & 3;
    const int hl = lane >> 4, l16 = lane & 15;
    const int kb = hl * 8;
    const int m0 = blockIdx.y * 128, n0 = blockIdx.x * 128;

    const int lr = t >> 3;         // 0..31 (4 row passes of 32)
    const int lc = (t & 7) * 8;    // 0..56

    v8f acc[4][2] = {};

    for (int k0 = 0; k0 < K; k0 += 64) {
        v8bf av[4], bv[4];
#pragma unroll
        for (int p = 0; p < 4; p++) {
            av[p] = *(const v8bf*)(A  + (size_t)(m0 + lr + p * 32) * K + k0 + lc);
            bv[p] = *(const v8bf*)(Bt + (size_t)(n0 + lr + p * 32) * K + k0 + lc);
        }
        if (k0 + 64 < K) {
            __builtin_prefetch(A  + (size_t)(m0 + lr) * K + k0 + 64 + lc, 0, 1);
            __builtin_prefetch(Bt + (size_t)(n0 + lr) * K + k0 + 64 + lc, 0, 1);
        }
#pragma unroll
        for (int p = 0; p < 4; p++) {
            *(v8bf*)&sA[lr + p * 32][lc] = av[p];
            *(v8bf*)&sB[lr + p * 32][lc] = bv[p];
        }
        __syncthreads();

#pragma unroll
        for (int kk = 0; kk < 64; kk += 32) {
            Frag16 af[4];
#pragma unroll
            for (int mt = 0; mt < 4; mt++) {
                int row = wm * 64 + mt * 16 + l16;
                af[mt].p[0] = *(const v8bf*)&sA[row][kk + kb];
                af[mt].p[1] = *(const v8bf*)&sA[row][kk + kb + 16];
            }
#pragma unroll
            for (int nt = 0; nt < 2; nt++) {
                int col = wn * 32 + nt * 16 + l16;
                Frag16 bfm;
                bfm.p[0] = *(const v8bf*)&sB[col][kk + kb];
                bfm.p[1] = *(const v8bf*)&sB[col][kk + kb + 16];
#pragma unroll
                for (int mt = 0; mt < 4; mt++)
                    acc[mt][nt] = __builtin_amdgcn_wmma_f32_16x16x32_bf16(
                        false, af[mt].v, false, bfm.v, (short)0, acc[mt][nt], false, false);
            }
        }
        __syncthreads();
    }

#pragma unroll
    for (int mt = 0; mt < 4; mt++)
#pragma unroll
        for (int nt = 0; nt < 2; nt++) {
            int col = n0 + wn * 32 + nt * 16 + l16;
            float bvv = (flags & FLAG_BIAS) ? bias[col] : 0.0f;
#pragma unroll
            for (int r = 0; r < 8; r++) {
                int row = m0 + wm * 64 + mt * 16 + hl * 8 + r;   // C layout: reg=row, lane=col
                float v = acc[mt][nt][r] + bvv;
                if (flags & FLAG_RES)     v += Res[(size_t)row * N + col];
                if (flags & FLAG_RELU)    v = fmaxf(v, 0.0f);
                if (flags & FLAG_OUTF32)  Cf[(size_t)row * N + col] = v;
                if (flags & FLAG_OUTBF16) Cb[(size_t)row * N + col] = (__bf16)v;
            }
        }
}

// ---------------------------------------------------------------------------
// Fused flash attention: one wave per (b, h, 16-query-row tile).
// scores = Q Kt (WMMA, direct global fragment loads), online softmax,
// P staged via LDS -> A fragments, P@V with pre-transposed Vt.
// ---------------------------------------------------------------------------
__global__ __launch_bounds__(32)
void attn_kernel(const __bf16* __restrict__ Q, const __bf16* __restrict__ K,
                 const __bf16* __restrict__ Vt, const int* __restrict__ tok,
                 __bf16* __restrict__ Out)
{
    const int lane = threadIdx.x;
    const int hl = lane >> 4, l16 = lane & 15;
    const int q0 = blockIdx.x * 16, h = blockIdx.y, b = blockIdx.z;
    __shared__ __bf16 sP[16][64];

    // Preload Q A-fragments for the two 32-wide K-dim steps over DH=64
    Frag16 aq[2];
    const __bf16* qbase = Q + ((size_t)(b * Sc + q0 + l16)) * Dc + h * DHc;
#pragma unroll
    for (int ks = 0; ks < 2; ks++) {
        int e0 = ks * 32 + hl * 8;
        aq[ks].p[0] = *(const v8bf*)(qbase + e0);
        aq[ks].p[1] = *(const v8bf*)(qbase + e0 + 16);
    }
    int qtok[8];
#pragma unroll
    for (int r = 0; r < 8; r++) qtok[r] = tok[b * Sc + q0 + hl * 8 + r];

    float m_i[8], l_i[8];
#pragma unroll
    for (int r = 0; r < 8; r++) { m_i[r] = -1e30f; l_i[r] = 0.0f; }
    v8f o[4] = {};

    const int kbmax = (q0 + 15) >> 6;      // causal: skip key blocks above diagonal
    for (int kblk = 0; kblk <= kbmax; kblk++) {
        const int key0 = kblk * 64;
        v8f sc[4] = {};
#pragma unroll
        for (int ks = 0; ks < 2; ks++)
#pragma unroll
            for (int nt = 0; nt < 4; nt++) {
                int key = key0 + nt * 16 + l16;
                const __bf16* kp = K + ((size_t)(b * Sc + key)) * Dc + h * DHc + ks * 32 + hl * 8;
                Frag16 bk;
                bk.p[0] = *(const v8bf*)kp;
                bk.p[1] = *(const v8bf*)(kp + 16);
                sc[nt] = __builtin_amdgcn_wmma_f32_16x16x32_bf16(
                    false, aq[ks].v, false, bk.v, (short)0, sc[nt], false, false);
            }
        // scale + causal/pad mask (C layout: reg r -> row q0+hl*8+r, lane -> key col)
#pragma unroll
        for (int nt = 0; nt < 4; nt++) {
            int key  = key0 + nt * 16 + l16;
            int ktok = tok[b * Sc + key];
#pragma unroll
            for (int r = 0; r < 8; r++) {
                int qrow = q0 + hl * 8 + r;
                float s  = sc[nt][r] * 0.125f;          // 1/sqrt(64)
                bool ok  = (key <= qrow) && (ktok != 0) && (qtok[r] != 0);
                sc[nt][r] = ok ? s : -1e9f;
            }
        }
        // online softmax (reductions across the 16 lanes holding a row)
#pragma unroll
        for (int r = 0; r < 8; r++) {
            float mx = sc[0][r];
#pragma unroll
            for (int nt = 1; nt < 4; nt++) mx = fmaxf(mx, sc[nt][r]);
            for (int off = 8; off >= 1; off >>= 1) mx = fmaxf(mx, __shfl_xor(mx, off, 32));
            float newm  = fmaxf(m_i[r], mx);
            float alpha = __expf(m_i[r] - newm);
            float rsum  = 0.0f;
#pragma unroll
            for (int nt = 0; nt < 4; nt++) {
                float p = __expf(sc[nt][r] - newm);
                sc[nt][r] = p;
                rsum += p;
            }
            for (int off = 8; off >= 1; off >>= 1) rsum += __shfl_xor(rsum, off, 32);
            l_i[r] = l_i[r] * alpha + rsum;
            m_i[r] = newm;
#pragma unroll
            for (int j = 0; j < 4; j++) o[j][r] *= alpha;
        }
        // stage P into LDS (C layout -> row-major), reload as A fragments
        __syncthreads();
#pragma unroll
        for (int nt = 0; nt < 4; nt++)
#pragma unroll
            for (int r = 0; r < 8; r++)
                sP[hl * 8 + r][nt * 16 + l16] = (__bf16)sc[nt][r];
        __syncthreads();
#pragma unroll
        for (int ks = 0; ks < 2; ks++) {
            Frag16 ap;
            ap.p[0] = *(const v8bf*)&sP[l16][ks * 32 + hl * 8];
            ap.p[1] = *(const v8bf*)&sP[l16][ks * 32 + hl * 8 + 16];
#pragma unroll
            for (int nt = 0; nt < 4; nt++) {
                int e = nt * 16 + l16;
                const __bf16* vp = Vt + ((size_t)((b * Hc + h) * DHc + e)) * Sc
                                      + key0 + ks * 32 + hl * 8;
                Frag16 bv;
                bv.p[0] = *(const v8bf*)vp;
                bv.p[1] = *(const v8bf*)(vp + 16);
                o[nt] = __builtin_amdgcn_wmma_f32_16x16x32_bf16(
                    false, ap.v, false, bv.v, (short)0, o[nt], false, false);
            }
        }
        __syncthreads();
    }
    // normalize + store bf16 output (head-interleaved [M, D])
#pragma unroll
    for (int r = 0; r < 8; r++) {
        float inv = 1.0f / l_i[r];
        int qrow = q0 + hl * 8 + r;
#pragma unroll
        for (int nt = 0; nt < 4; nt++)
            Out[((size_t)(b * Sc + qrow)) * Dc + h * DHc + nt * 16 + l16] =
                (__bf16)(o[nt][r] * inv);
    }
}

// ---------------------------------------------------------------------------
// Host orchestration
// ---------------------------------------------------------------------------
extern "C" void kernel_launch(void* const* d_in, const int* in_sizes, int n_in,
                              void* d_out, int out_size, void* d_ws, size_t ws_size,
                              hipStream_t stream)
{
    const int*   tok = (const int*)d_in[0];
    const float* E   = (const float*)d_in[1];
    const float* P   = (const float*)d_in[2];
    const float* Wq  = (const float*)d_in[3];
    const float* bq  = (const float*)d_in[4];
    const float* Wk  = (const float*)d_in[5];
    const float* bk  = (const float*)d_in[6];
    const float* Wv  = (const float*)d_in[7];
    const float* bv  = (const float*)d_in[8];
    const float* W0  = (const float*)d_in[9];
    const float* b0  = (const float*)d_in[10];
    const float* g1  = (const float*)d_in[11];
    const float* be1 = (const float*)d_in[12];
    const float* g2  = (const float*)d_in[13];
    const float* be2 = (const float*)d_in[14];
    const float* W1  = (const float*)d_in[15];
    const float* b1  = (const float*)d_in[16];
    const float* W2  = (const float*)d_in[17];
    const float* b2  = (const float*)d_in[18];
    float* out = (float*)d_out;

    char* wp = (char*)d_ws;
    auto take = [&](size_t bytes) -> char* {
        char* p = wp; wp += (bytes + 255) & ~(size_t)255; return p;
    };
    float*  X     = (float*) take((size_t)Mc * Dc * 4);
    __bf16* hbf   = (__bf16*)take((size_t)Mc * Dc * 2);
    __bf16* Qb    = (__bf16*)take((size_t)Mc * Dc * 2);
    __bf16* Kb    = (__bf16*)take((size_t)Mc * Dc * 2);
    __bf16* Vb    = (__bf16*)take((size_t)Mc * Dc * 2);
    __bf16* Vt    = (__bf16*)take((size_t)Mc * Dc * 2);
    __bf16* attnb = (__bf16*)take((size_t)Mc * Dc * 2);
    __bf16* ffb   = (__bf16*)take((size_t)Mc * DFFc * 2);
    __bf16* Wqp   = (__bf16*)take((size_t)Dc * Dc * 2);      // [N=D][K=D]
    __bf16* Wkp   = (__bf16*)take((size_t)Dc * Dc * 2);
    __bf16* Wvp   = (__bf16*)take((size_t)Dc * Dc * 2);
    __bf16* W0p   = (__bf16*)take((size_t)Dc * Dc * 2);
    __bf16* W1p   = (__bf16*)take((size_t)Dc * DFFc * 2);    // [DFF][D]
    __bf16* W2p   = (__bf16*)take((size_t)DFFc * Dc * 2);    // [D][DFF]
    __bf16* Ebf   = (__bf16*)take((size_t)Vc * Dc * 2);      // E is already [N=V][K=D]

    const dim3 blk256(256);
    // X = E[tokens] + P
    embed_kernel<<<dim3((Mc * Dc) / 256), blk256, 0, stream>>>(tok, E, P, X);
    // Tied head B-operand: Bt = E (already [V, D]); just convert to bf16
    cvt_kernel<<<dim3((Vc * Dc) / 256), blk256, 0, stream>>>(E, Ebf, Vc * Dc);

    for (int l = 0; l < Lc; l++) {
        const size_t wqkv_off = (size_t)l * Hc * Dc * DHc;
        // h = LN1(X)
        ln_kernel<<<dim3(Mc), blk256, 0, stream>>>(X, g1 + l * Dc, be1 + l * Dc, hbf);
        // QKV weights [H,D,DH] -> transposed-B [N=H*DH][K=D] bf16 (per-head batch)
        transpose_cvt_kernel<<<dim3(DHc / 32, Dc / 32, Hc), blk256, 0, stream>>>(
            Wq + wqkv_off, Wqp, DHc, Dc, (long)Dc * DHc, (long)DHc * Dc);
        transpose_cvt_kernel<<<dim3(DHc / 32, Dc / 32, Hc), blk256, 0, stream>>>(
            Wk + wqkv_off, Wkp, DHc, Dc, (long)Dc * DHc, (long)DHc * Dc);
        transpose_cvt_kernel<<<dim3(DHc / 32, Dc / 32, Hc), blk256, 0, stream>>>(
            Wv + wqkv_off, Wvp, DHc, Dc, (long)Dc * DHc, (long)DHc * Dc);
        // Q/K/V projections
        gemm_kernel<<<dim3(Dc / 128, Mc / 128), blk256, 0, stream>>>(
            hbf, Wqp, bq + l * Dc, nullptr, nullptr, Qb, Mc, Dc, Dc, FLAG_BIAS | FLAG_OUTBF16);
        gemm_kernel<<<dim3(Dc / 128, Mc / 128), blk256, 0, stream>>>(
            hbf, Wkp, bk + l * Dc, nullptr, nullptr, Kb, Mc, Dc, Dc, FLAG_BIAS | FLAG_OUTBF16);
        gemm_kernel<<<dim3(Dc / 128, Mc / 128), blk256, 0, stream>>>(
            hbf, Wvp, bv + l * Dc, nullptr, nullptr, Vb, Mc, Dc, Dc, FLAG_BIAS | FLAG_OUTBF16);
        // Vt = V transposed per head (tiled)
        transpose_v_kernel<<<dim3(DHc / 32, Sc / 32, Bc * Hc), blk256, 0, stream>>>(Vb, Vt);
        // fused flash attention
        attn_kernel<<<dim3(Sc / 16, Hc, Bc), dim3(32), 0, stream>>>(Qb, Kb, Vt, tok, attnb);
        // X = attn @ W0 + b0 + X   (W0 [D,D] -> Bt [D,D] transposed)
        transpose_cvt_kernel<<<dim3(Dc / 32, Dc / 32, 1), blk256, 0, stream>>>(
            W0 + (size_t)l * Dc * Dc, W0p, Dc, Dc, 0L, 0L);
        gemm_kernel<<<dim3(Dc / 128, Mc / 128), blk256, 0, stream>>>(
            attnb, W0p, b0 + l * Dc, X, X, nullptr, Mc, Dc, Dc,
            FLAG_BIAS | FLAG_RES | FLAG_OUTF32);
        // h = LN2(X)
        ln_kernel<<<dim3(Mc), blk256, 0, stream>>>(X, g2 + l * Dc, be2 + l * Dc, hbf);
        // FFN: W1 [D,DFF] -> Bt [DFF,D]; W2 [DFF,D] -> Bt [D,DFF]
        transpose_cvt_kernel<<<dim3(DFFc / 32, Dc / 32, 1), blk256, 0, stream>>>(
            W1 + (size_t)l * Dc * DFFc, W1p, DFFc, Dc, 0L, 0L);
        gemm_kernel<<<dim3(DFFc / 128, Mc / 128), blk256, 0, stream>>>(
            hbf, W1p, b1 + l * DFFc, nullptr, nullptr, ffb, Mc, DFFc, Dc,
            FLAG_BIAS | FLAG_RELU | FLAG_OUTBF16);
        transpose_cvt_kernel<<<dim3(Dc / 32, DFFc / 32, 1), blk256, 0, stream>>>(
            W2 + (size_t)l * DFFc * Dc, W2p, Dc, DFFc, 0L, 0L);
        gemm_kernel<<<dim3(Dc / 128, Mc / 128), blk256, 0, stream>>>(
            ffb, W2p, b2 + l * Dc, X, X, nullptr, Mc, Dc, DFFc,
            FLAG_BIAS | FLAG_RES | FLAG_OUTF32);
    }
    // logits = X @ E^T  (Bt = E itself)
    cvt_kernel<<<dim3((Mc * Dc) / 256), blk256, 0, stream>>>(X, hbf, Mc * Dc);
    gemm_kernel<<<dim3(Vc / 128, Mc / 128), blk256, 0, stream>>>(
        hbf, Ebf, nullptr, nullptr, out, nullptr, Mc, Vc, Dc, FLAG_OUTF32);
}
